// Encoder_8564164788234
// MI455X (gfx1250) — matrix operand order repaired and checked by reading, not compiled, not run on hardware
//
#include <hip/hip_runtime.h>
#include <hip/hip_bf16.h>

// ---------------------------------------------------------------------------
// pix2pixHD encoder on gfx1250.
//  - Activations: NHWC f16 with a 32-f16 zeroed header per buffer; predicates
//    fold into a 32-bit offset select (0 -> zero header), base+voffset loads.
//  - Each lane's WMMA B fragment: one contiguous 32B load (fast layers) or
//    4x8B tap loads (head, channels padded 3->4, zeros baked into packed A).
//  - Weights pre-packed per layer into the exact v_wmma A-fragment layout.
//  - Convs: implicit GEMM on v_wmma_f32_16x16x32_f16 (f32 accumulate).
//  - Raw conv out: NHWC f32 -> BN(train) + ReLU -> f16 NHWC.
//  - Tail: tanh + per-(batch, instance-id) scatter-mean -> NCHW f32 output.
// ---------------------------------------------------------------------------

typedef _Float16 f16;
typedef f16   v16h __attribute__((ext_vector_type(16)));
typedef f16   v4h  __attribute__((ext_vector_type(4)));
typedef float v8f  __attribute__((ext_vector_type(8)));

#define WV 4   // waves per block
#define NT 4   // 16-pixel tiles per wave (A-fragment reuse)

__device__ __forceinline__ int reflect_idx(int t, int n) {
    t = (t < 0) ? -t : t;
    t = (t >= n) ? (2 * n - 2 - t) : t;
    return t;
}

// ---------------------------------------------------------------------------
// Weight packing into WMMA A-fragment order, f16.
// CIN = K-layout channel count (may be padded); CINW = real weight channels.
// ORD=0: k = tap*CIN + ci (tap-major). MODE 2 = convT weights (Cin,Cout,kh,kw).
// A elem map: lane l (M=l&15, half=l>>4), e -> kt*32+(half<<4)+e
// ---------------------------------------------------------------------------
template <int CIN, int CINW, int COUT, int KHW, int MODE>
__global__ void pack_weights(const float* __restrict__ W, f16* __restrict__ Wp)
{
    constexpr int KK   = KHW * KHW;
    constexpr int Kdim = CIN * KK;
    constexpr int NKT  = (Kdim + 31) / 32;
    constexpr int NOT  = (COUT + 15) / 16;
    int idx = blockIdx.x * blockDim.x + threadIdx.x;   // one per (ot, kt, lane)
    if (idx >= NOT * NKT * 32) return;
    int lane = idx & 31;
    int kt   = (idx >> 5) % NKT;
    int ot   = (idx >> 5) / NKT;
    int l16 = lane & 15, half = lane >> 4;
    int m = ot * 16 + l16;
    v16h vals;
#pragma unroll
    for (int e = 0; e < 16; ++e) {
        int k = kt * 32 + (half << 4) + e;
        int tap = k / CIN, ci = k % CIN;
        float v = 0.f;
        if (k < Kdim && m < COUT && ci < CINW)
            v = (MODE == 2) ? W[((long)ci * COUT + m) * KK + tap]
                            : W[((long)m * CINW + ci) * KK + tap];
        vals[e] = (f16)v;
    }
    *(v16h*)(Wp + (long)idx * 16) = vals;
}

// ---------------------------------------------------------------------------
// Fast conv: NHWC f16 in (X = buffer base; data at +32, zeros at 0..31),
// NHWC f32 out, pre-packed weights. Requires CIN % 32 == 0.
// MODE: 0 zero pad, 1 reflect, 2 ConvT(k3,s2,p1,op1)
// ---------------------------------------------------------------------------
template <int CIN, int COUT, int KHW, int STRIDE, int PAD, int MODE>
__global__ void __launch_bounds__(WV * 32)
conv2d_wmma_nhwc(const f16* __restrict__ X, const f16* __restrict__ Wp,
                 const float* __restrict__ Bs, float* __restrict__ Y,
                 int Bn, int Hin, int Win, int Ho, int Wo)
{
    constexpr int KK   = KHW * KHW;
    constexpr int Kdim = CIN * KK;
    constexpr int NKT  = Kdim / 32;
    static_assert(CIN % 32 == 0, "fast path needs CIN % 32 == 0");

    const int lane = threadIdx.x & 31;
    const int wave = threadIdx.x >> 5;
    const int half = lane >> 4;
    const int l16  = lane & 15;
    const int oc0  = blockIdx.y * 16;
    const int HW   = Ho * Wo;
    const long Npix = (long)Bn * HW;
    const long n0   = ((long)blockIdx.x * WV + wave) * (NT * 16);

    int py[NT], px[NT];
    unsigned tb[NT];
    bool pv[NT];
#pragma unroll
    for (int t = 0; t < NT; ++t) {
        long p = n0 + t * 16 + l16;
        pv[t] = (p < Npix);
        long pc = pv[t] ? p : 0;
        int pb = (int)(pc / HW);
        int r  = (int)(pc % HW);
        py[t] = r / Wo;
        px[t] = r % Wo;
        tb[t] = 32u + (unsigned)pb * (unsigned)(Hin * Win * CIN);
    }

    v8f acc[NT];
#pragma unroll
    for (int t = 0; t < NT; ++t)
#pragma unroll
        for (int i = 0; i < 8; ++i) acc[t][i] = 0.f;

    unsigned woff = ((unsigned)blockIdx.y * NKT * 32 + lane) * 16;

    for (int kt = 0; kt < NKT; ++kt) {
        v16h afrag = *(const v16h*)(Wp + woff);
        woff += 32 * 16;
        __builtin_prefetch(Wp + woff, 0, 1);       // prefetch next A tile
        const int k0  = kt * 32;
        const int tap = k0 / CIN;                  // constant shift (CIN pow2)
        const int ci0 = (k0 - tap * CIN) + (half << 4);
        const int kh  = tap / KHW, kw = tap % KHW;
#pragma unroll
        for (int t = 0; t < NT; ++t) {
            int iy, ix;
            bool ok = pv[t];
            if (MODE == 2) {                        // transposed-conv gather
                int ty = py[t] + 1 - kh, tx = px[t] + 1 - kw;
                ok = ok && (ty >= 0) && (tx >= 0) && !(ty & 1) && !(tx & 1);
                iy = ty >> 1; ix = tx >> 1;
                ok = ok && (iy < Hin) && (ix < Win);
            } else if (MODE == 1) {                 // reflect pad, in range
                iy = reflect_idx(py[t] * STRIDE - PAD + kh, Hin);
                ix = reflect_idx(px[t] * STRIDE - PAD + kw, Win);
            } else {                                // zero pad
                iy = py[t] * STRIDE - PAD + kh;
                ix = px[t] * STRIDE - PAD + kw;
                ok = ok && (iy >= 0) && (iy < Hin) && (ix >= 0) && (ix < Win);
            }
            // 32-bit offset; 0 -> zeroed buffer header when invalid
            unsigned off = tb[t] + (unsigned)((iy * Win + ix) * CIN + ci0);
            off = ok ? off : 0u;
            v16h bfrag = *(const v16h*)(X + off);
            acc[t] = __builtin_amdgcn_wmma_f32_16x16x32_f16(
                false, afrag, false, bfrag, (short)0, acc[t], false, false);
        }
    }

    // store: lane column = pixel, 8 consecutive out channels per lane
    const int occ = oc0 + (half << 3);
#pragma unroll
    for (int t = 0; t < NT; ++t) {
        if (!pv[t]) continue;
        unsigned yoff = (unsigned)(n0 + t * 16 + l16) * (unsigned)COUT + occ;
        if (COUT % 16 == 0) {
            v8f bias = *(const v8f*)(Bs + occ);
            v8f outv;
#pragma unroll
            for (int r = 0; r < 8; ++r) outv[r] = acc[t][r] + bias[r];
            *(v8f*)(Y + yoff) = outv;
        } else {
#pragma unroll
            for (int r = 0; r < 8; ++r) {
                if (occ + r < COUT) Y[yoff + r] = acc[t][r] + Bs[occ + r];
            }
        }
    }
}

// ---------------------------------------------------------------------------
// Head: NHWC f16 input padded to C=4, tap-major K (k = tap*4 + ci), Kdim=196.
// Zero padding (K tail + 4th channel) is baked into packed A weights, so B
// needs no value selects; per-tap 8B vector loads.
// ---------------------------------------------------------------------------
__global__ void __launch_bounds__(WV * 32)
conv2d_wmma_head(const f16* __restrict__ X4, const f16* __restrict__ Wp,
                 const float* __restrict__ Bs, float* __restrict__ Y, int Bn)
{
    constexpr int H = 512, Wd = 512, HW = H * Wd;
    constexpr int NKT = 7, COUT = 32, PAD = 3;

    const int lane = threadIdx.x & 31;
    const int wave = threadIdx.x >> 5;
    const int half = lane >> 4;
    const int l16  = lane & 15;
    const int oc0  = blockIdx.y * 16;
    const long Npix = (long)Bn * HW;
    const long n0   = ((long)blockIdx.x * WV + wave) * (NT * 16);

    int py[NT], px[NT], tbp[NT];
    bool pv[NT];
#pragma unroll
    for (int t = 0; t < NT; ++t) {
        long p = n0 + t * 16 + l16;
        pv[t] = (p < Npix);
        long pc = pv[t] ? p : 0;
        int pb = (int)(pc / HW);
        int r  = (int)(pc % HW);
        py[t] = r / Wd;
        px[t] = r % Wd;
        tbp[t] = pb * HW;
    }

    v8f acc[NT];
#pragma unroll
    for (int t = 0; t < NT; ++t)
#pragma unroll
        for (int i = 0; i < 8; ++i) acc[t][i] = 0.f;

    unsigned woff = ((unsigned)blockIdx.y * NKT * 32 + lane) * 16;

    for (int kt = 0; kt < NKT; ++kt) {
        v16h afrag = *(const v16h*)(Wp + woff);
        woff += 32 * 16;
        const int tap0 = (kt * 32 + (half << 4)) >> 2;
#pragma unroll
        for (int t = 0; t < NT; ++t) {
            v16h bfrag;
#pragma unroll
            for (int j = 0; j < 4; ++j) {
                int tap = tap0 + j;
                tap = (tap < 49) ? tap : 0;         // A is zero there anyway
                int kh = tap / 7, kw = tap % 7;
                int iy = reflect_idx(py[t] - PAD + kh, H);
                int ix = reflect_idx(px[t] - PAD + kw, Wd);
                unsigned off = ((unsigned)(tbp[t] + (iy << 9) + ix)) << 2;
                v4h q = *(const v4h*)(X4 + off);
                bfrag[4 * j + 0] = q[0];
                bfrag[4 * j + 1] = q[1];
                bfrag[4 * j + 2] = q[2];
                bfrag[4 * j + 3] = q[3];
            }
            acc[t] = __builtin_amdgcn_wmma_f32_16x16x32_f16(
                false, afrag, false, bfrag, (short)0, acc[t], false, false);
        }
    }

    const int occ = oc0 + (half << 3);
    v8f bias = *(const v8f*)(Bs + occ);
#pragma unroll
    for (int t = 0; t < NT; ++t) {
        if (!pv[t]) continue;
        unsigned yoff = (unsigned)(n0 + t * 16 + l16) * (unsigned)COUT + occ;
        v8f outv;
#pragma unroll
        for (int r = 0; r < 8; ++r) outv[r] = acc[t][r] + bias[r];
        *(v8f*)(Y + yoff) = outv;
    }
}

// ---------------------------------------------------------------------------
__global__ void zero_f32(float* __restrict__ p, int n) {
    int i = blockIdx.x * blockDim.x + threadIdx.x;
    if (i < n) p[i] = 0.f;
}

// NCHW f32 (C=3) -> NHWC f16 padded to C=4
__global__ void nchw3_to_nhwc4_f16(const float* __restrict__ X, f16* __restrict__ O,
                                   int Bn, int HW)
{
    long N = (long)Bn * HW * 4;
    long i = (long)blockIdx.x * blockDim.x + threadIdx.x;
    if (i >= N) return;
    int c   = (int)(i & 3);
    long bp = i >> 2;
    int b   = (int)(bp / HW);
    int pix = (int)(bp % HW);
    O[i] = (c < 3) ? (f16)X[((long)b * 3 + c) * HW + pix] : (f16)0.f;
}

// per-channel sum/sumsq over all pixels, NHWC. CG = channels per block group.
__global__ void bn_reduce_nhwc(const float* __restrict__ X, float* __restrict__ stat,
                               long NP, int C, int CG)
{
    const int tid = threadIdx.x;
    const int cg  = blockIdx.y * CG;
    const int cl  = tid % CG;
    const int row = tid / CG;
    const int rows = 256 / CG;
    float sum = 0.f, sq = 0.f;
    for (long p = (long)blockIdx.x * rows + row; p < NP; p += (long)gridDim.x * rows) {
        float v = X[p * C + cg + cl];
        sum += v; sq += v * v;
    }
    __shared__ float s1[256], s2[256];
    s1[tid] = sum; s2[tid] = sq;
    __syncthreads();
    for (int step = 128; step >= CG; step >>= 1) {
        if (tid < step) { s1[tid] += s1[tid + step]; s2[tid] += s2[tid + step]; }
        __syncthreads();
    }
    if (tid < CG) {
        atomicAdd(&stat[cg + tid], s1[tid]);
        atomicAdd(&stat[512 + cg + tid], s2[tid]);
    }
}

// normalize + ReLU, NHWC f32 -> NHWC f16
__global__ void bn_apply_relu_f16(const float* __restrict__ X, f16* __restrict__ O,
                                  const float* __restrict__ stat,
                                  const float* __restrict__ g, const float* __restrict__ beta,
                                  int Cmask, long N, float invCnt)
{
    long i = (long)blockIdx.x * blockDim.x + threadIdx.x;
    if (i >= N) return;
    int c = (int)(i & Cmask);                 // C is a power of two
    float m   = stat[c] * invCnt;
    float var = stat[512 + c] * invCnt - m * m;
    float y = g[c] * (X[i] - m) * rsqrtf(var + 1e-5f) + beta[c];
    O[i] = (f16)((y > 0.f) ? y : 0.f);
}

// tail: tanh in place (NHWC f32, C=3) + instance scatter-add
__global__ void tanh_inst_accum(float* __restrict__ Yf, const int* __restrict__ inst,
                                float* __restrict__ isum, float* __restrict__ icnt,
                                int Bn, int HW, int Kinst)
{
    long N = (long)Bn * HW * 3;
    long i = (long)blockIdx.x * blockDim.x + threadIdx.x;
    if (i >= N) return;
    int b   = (int)(i / ((long)HW * 3));
    int rem = (int)(i - (long)b * HW * 3);
    int pix = rem / 3, c = rem % 3;
    float t = tanhf(Yf[i]);
    Yf[i] = t;
    int seg = b * Kinst + inst[(long)b * HW + pix];
    atomicAdd(&isum[c * (8 * 32) + seg], t);
    if (c == 0) atomicAdd(&icnt[seg], 1.0f);
}

// broadcast means into NCHW f32 output
__global__ void inst_broadcast(const int* __restrict__ inst,
                               const float* __restrict__ isum, const float* __restrict__ icnt,
                               float* __restrict__ out, int Bn, int HW, int Kinst)
{
    long N = (long)Bn * 3 * HW;
    long i = (long)blockIdx.x * blockDim.x + threadIdx.x;
    if (i >= N) return;
    int b   = (int)(i / ((long)3 * HW));
    int c   = (int)((i / HW) % 3);
    int pix = (int)(i % HW);
    int seg = b * Kinst + inst[(long)b * HW + pix];
    out[i] = isum[c * (8 * 32) + seg] / fmaxf(icnt[seg], 1.0f);
}

// ---------------------------------------------------------------------------
struct LD { int Cin, Hin, Cout, Ho; };
static const LD g_L[10] = {
    {  3, 512,  32, 512},   // head: reflect 7x7 (C->4 padded path)
    { 32, 512,  64, 256},   // down 1
    { 64, 256, 128, 128},   // down 2
    {128, 128, 256,  64},   // down 3
    {256,  64, 512,  32},   // down 4
    {512,  32, 256,  64},   // up 1 (convT)
    {256,  64, 128, 128},   // up 2
    {128, 128,  64, 256},   // up 3
    { 64, 256,  32, 512},   // up 4
    { 32, 512,   3, 512},   // tail: reflect 7x7
};
// packed-weight sizes in f16 elements: NOT * NKT * 512
static const long g_wpsz[10] = { 7168, 18432, 73728, 294912, 1179648,
                                 1179648, 294912, 73728, 18432, 25088 };

static dim3 conv_grid(int l) {
    const LD d = g_L[l];
    const long Npix = (long)8 * d.Ho * d.Ho;
    const int pixPerBlk = 16 * WV * NT;
    return dim3((unsigned)((Npix + pixPerBlk - 1) / pixPerBlk),
                (unsigned)((d.Cout + 15) / 16));
}

extern "C" void kernel_launch(void* const* d_in, const int* in_sizes, int n_in,
                              void* d_out, int out_size, void* d_ws, size_t ws_size,
                              hipStream_t stream)
{
    const float* X0 = (const float*)d_in[0];
    const int* inst = (const int*)d_in[1];

    // parameter pointers (jax-pytree sorted order vs insertion order)
    const float *cw[10], *cb[10], *bg[9], *bb[9];
    if (in_sizes[2] == 32) {  // sorted: bn{b,g}x9 then conv{b,w}x10
        for (int i = 0; i < 9; ++i) {
            bb[i] = (const float*)d_in[2 + 2 * i];
            bg[i] = (const float*)d_in[3 + 2 * i];
        }
        for (int j = 0; j < 10; ++j) {
            cb[j] = (const float*)d_in[20 + 2 * j];
            cw[j] = (const float*)d_in[21 + 2 * j];
        }
    } else {                  // insertion: conv{w,b}x10 then bn{g,b}x9
        for (int j = 0; j < 10; ++j) {
            cw[j] = (const float*)d_in[2 + 2 * j];
            cb[j] = (const float*)d_in[3 + 2 * j];
        }
        for (int i = 0; i < 9; ++i) {
            bg[i] = (const float*)d_in[22 + 2 * i];
            bb[i] = (const float*)d_in[23 + 2 * i];
        }
    }

    // workspace layout (act buffers have 32-f16 zeroed headers)
    float* raw   = (float*)d_ws;                  // 67,108,864 f32 (NHWC conv out)
    f16*  actA   = (f16*)(raw + 67108864);        // 32 + 67,108,864 f16
    f16*  actB   = actA + 67108896;               // 32 + 67,108,864 f16
    float* stat  = (float*)(actB + 67108896);     // 1024 f32
    float* isum  = stat + 1024;                   // 768 f32
    float* icnt  = isum + 768;                    // 256 f32
    f16*  wpk    = (f16*)(icnt + 256);            // packed weights, ~3.2M f16
    f16*  wp[10];
    {
        long off = 0;
        for (int l = 0; l < 10; ++l) { wp[l] = wpk + off; off += g_wpsz[l]; }
    }
    f16* x4 = wpk + 3165696;                      // NHWC4 f16 input, 8,388,608 f16

    const int Bn = 8, Kinst = 32;
    const int HW0 = 512 * 512;
    dim3 blk(WV * 32);

    // ---- init act-buffer zero headers + convert input to NHWC4 f16 ----
    zero_f32<<<1, 32, 0, stream>>>((float*)actA, 16);
    zero_f32<<<1, 32, 0, stream>>>((float*)actB, 16);
    {
        long N = (long)Bn * HW0 * 4;
        nchw3_to_nhwc4_f16<<<(unsigned)((N + 255) / 256), 256, 0, stream>>>(X0, x4, Bn, HW0);
    }

    // ---- pack all layer weights into WMMA fragment layout ----
#define PACK(l, CI, CIW, CO, KH, MD) \
    { long n = g_wpsz[l] / 16; \
      pack_weights<CI, CIW, CO, KH, MD><<<(unsigned)((n + 255) / 256), 256, 0, stream>>>(cw[l], wp[l]); }
    PACK(0,   4,   3,  32, 7, 0)
    PACK(1,  32,  32,  64, 3, 0) PACK(2,  64,  64, 128, 3, 0) PACK(3, 128, 128, 256, 3, 0)
    PACK(4, 256, 256, 512, 3, 0) PACK(5, 512, 512, 256, 3, 2) PACK(6, 256, 256, 128, 3, 2)
    PACK(7, 128, 128,  64, 3, 2) PACK(8,  64,  64,  32, 3, 2) PACK(9,  32,  32,   3, 7, 0)
#undef PACK

    // ---- per-layer BN helper (writes f16 data at actOut base + 32) ----
    auto bn = [&](int l, int C, int HO, f16* actOutBase) {
        const long NP = (long)Bn * HO * HO;
        zero_f32<<<4, 256, 0, stream>>>(stat, 1024);
        const int CG = (C < 256) ? C : 256;
        dim3 rg(64, (unsigned)(C / CG));
        bn_reduce_nhwc<<<rg, 256, 0, stream>>>(raw, stat, NP, C, CG);
        const long N = NP * C;
        const float invCnt = 1.0f / (float)NP;
        bn_apply_relu_f16<<<(unsigned)((N + 255) / 256), 256, 0, stream>>>(
            raw, actOutBase + 32, stat, bg[l], bb[l], C - 1, N, invCnt);
    };

    // ---- layer 0 (head) ----
    conv2d_wmma_head<<<conv_grid(0), blk, 0, stream>>>(x4, wp[0], cb[0], raw, Bn);
    bn(0, 32, 512, actA);

    // ---- layers 1..8 (fast NHWC path, ping-pong actA/actB) ----
    f16* aIn = actA;
    f16* aOut = actB;
#define FCONV(l, CI, CO, KH, ST, PD, MD) \
    conv2d_wmma_nhwc<CI, CO, KH, ST, PD, MD><<<conv_grid(l), blk, 0, stream>>>( \
        aIn, wp[l], cb[l], raw, Bn, g_L[l].Hin, g_L[l].Hin, g_L[l].Ho, g_L[l].Ho); \
    bn(l, CO, g_L[l].Ho, aOut); \
    { f16* tmp = aIn; aIn = aOut; aOut = tmp; }
    FCONV(1,  32,  64, 3, 2, 1, 0)
    FCONV(2,  64, 128, 3, 2, 1, 0)
    FCONV(3, 128, 256, 3, 2, 1, 0)
    FCONV(4, 256, 512, 3, 2, 1, 0)
    FCONV(5, 512, 256, 3, 2, 1, 2)
    FCONV(6, 256, 128, 3, 2, 1, 2)
    FCONV(7, 128,  64, 3, 2, 1, 2)
    FCONV(8,  64,  32, 3, 2, 1, 2)
#undef FCONV

    // ---- layer 9 (tail conv, Cout=3, NHWC f32 raw) ----
    conv2d_wmma_nhwc<32, 3, 7, 1, 3, 1><<<conv_grid(9), blk, 0, stream>>>(
        aIn, wp[9], cb[9], raw, Bn, 512, 512, 512, 512);

    // ---- tanh + instance mean ----
    {
        const long N = (long)Bn * 3 * HW0;
        zero_f32<<<4, 256, 0, stream>>>(isum, 768 + 256);
        tanh_inst_accum<<<(unsigned)((N + 255) / 256), 256, 0, stream>>>(
            raw, inst, isum, icnt, Bn, HW0, Kinst);
        inst_broadcast<<<(unsigned)((N + 255) / 256), 256, 0, stream>>>(
            inst, isum, icnt, (float*)d_out, Bn, HW0, Kinst);
    }
    (void)n_in; (void)out_size; (void)ws_size;
}
